// CMAPFeatureExtractor_34883724378760
// MI455X (gfx1250) — matrix operand order repaired and checked by reading, not compiled, not run on hardware
//
#include <hip/hip_runtime.h>
#include <hip/hip_bf16.h>
#include <stdint.h>

// ---------------------------------------------------------------------------
// CDNA5 (gfx1250) wave32 WMMA bf16 pipeline for the CMAP feature extractor.
// Heavy math: v_wmma_f32_16x16x32_bf16. Tile feed: TDM tensor_load_to_lds
// (async, TENSORcnt-tracked, LDS double-buffered) when the builtin exists.
// ---------------------------------------------------------------------------

typedef __attribute__((ext_vector_type(16))) __bf16 v16bf;
typedef __attribute__((ext_vector_type(8)))  float  v8f;
typedef __attribute__((ext_vector_type(4)))  unsigned int u32x4;
typedef __attribute__((ext_vector_type(8)))  int i32x8;
typedef __attribute__((ext_vector_type(4)))  int i32x4;

#ifdef __has_builtin
#  if __has_builtin(__builtin_amdgcn_tensor_load_to_lds)
#    define HAVE_TDM 1
#  else
#    define HAVE_TDM 0
#  endif
#  if __has_builtin(__builtin_amdgcn_s_wait_tensorcnt)
#    define HAVE_WAIT_TENSOR 1
#  else
#    define HAVE_WAIT_TENSOR 0
#  endif
#else
#  define HAVE_TDM 0
#  define HAVE_WAIT_TENSOR 0
#endif

struct __align__(16) U16x8 { uint32_t w[4]; };   // 8 bf16 = 16 bytes
union FragBF { v16bf v; U16x8 h[2]; };

__device__ __forceinline__ uint16_t f2bf(float f) {
  uint32_t u = __float_as_uint(f);
  u += 0x7FFFu + ((u >> 16) & 1u);               // round-to-nearest-even
  return (uint16_t)(u >> 16);
}
__device__ __forceinline__ float bf2f(uint16_t h) {
  return __uint_as_float(((uint32_t)h) << 16);
}

template <int CNT>
__device__ __forceinline__ void wait_tensorcnt() {
#if HAVE_WAIT_TENSOR
  __builtin_amdgcn_s_wait_tensorcnt(CNT);
#elif HAVE_TDM
  if (CNT == 0) asm volatile("s_wait_tensorcnt 0x0" ::: "memory");
  else          asm volatile("s_wait_tensorcnt 0x2" ::: "memory");
#endif
}

#define BM 128
#define BN 128
#define BK 32
#define LDSW 40   // u16 row stride in LDS (80B = 64B tile row + 16B bank pad)

// Issue one TDM load of a [128 rows x 32 bf16] tile into LDS with the padded
// row stride (pad_enable: +4 DWORDs after every 16 DWORDs => 80B rows).
__device__ __forceinline__ void tdm_load_tile(uint32_t lds_byte_addr,
                                              const uint16_t* gsrc,   // tile start
                                              int tensor_d0,          // row elems from tile start
                                              int tensor_d1,          // rows
                                              int stride_elems)       // row stride (elems)
{
#if HAVE_TDM
  const uint64_t ga = (uint64_t)(uintptr_t)gsrc;
  u32x4 g0;
  g0[0] = 1u;                                        // count=1 (valid user D#)
  g0[1] = lds_byte_addr;                             // lds_addr
  g0[2] = (uint32_t)ga;                              // global_addr[31:0]
  g0[3] = (uint32_t)((ga >> 32) & 0x1FFFFFFull)      // global_addr[56:32]
          | (2u << 30);                              // type=2 ("image")
  i32x8 g1;
  g1[0] = (int)((1u << 16)                           // data_size = 2 bytes
                | (1u << 20)                         // pad_enable
                | (3u << 22)                         // pad_interval: 16 DWORDs
                | (3u << 25));                       // pad_amount: 4 DWORDs
  g1[1] = (int)(((uint32_t)tensor_d0 & 0xFFFFu) << 16);               // dim0 lo
  g1[2] = (int)(((uint32_t)tensor_d0 >> 16)
                | (((uint32_t)tensor_d1 & 0xFFFFu) << 16));           // dim0 hi | dim1 lo
  g1[3] = (int)(((uint32_t)tensor_d1 >> 16) | (32u << 16));           // dim1 hi | tile_dim0=32
  g1[4] = (int)128u;                                                  // tile_dim1=128, tile_dim2=0
  g1[5] = (int)(uint32_t)stride_elems;                                // dim0_stride lo32
  g1[6] = 0;                                                          // stride hi | dim1_stride lo
  g1[7] = 0;
  const i32x4 z4 = {0, 0, 0, 0};
#  if __has_include(<hip/amd_detail/amd_gfx1250_TDM.h>)
  const i32x8 z8 = {0, 0, 0, 0, 0, 0, 0, 0};
  __builtin_amdgcn_tensor_load_to_lds(g0, g1, z4, z4, z8, 0);
#  else
  __builtin_amdgcn_tensor_load_to_lds(g0, g1, z4, z4, 0);
#  endif
#else
  (void)lds_byte_addr; (void)gsrc; (void)tensor_d0; (void)tensor_d1; (void)stride_elems;
#endif
}

// One BK=32 step: build fragments per CDNA5 16-bit VGPR layouts and do 8 WMMAs.
// A (16x32): lane L holds row M=L&15; lanes 0-15 K={0..7,16..23}, 16-31 K={8..15,24..31}.
// B (32x16): lane L holds col N=L&15; lanes 0-15 K={0..7,16..23}, 16-31 K={8..15,24..31}.
__device__ __forceinline__ void compute_tile(const uint16_t* __restrict__ lAp,
                                             const uint16_t* __restrict__ lBp,
                                             v8f (&acc)[2][4],
                                             int wmi, int wni, int l15, int hi)
{
  FragBF a[2], b[4];
#pragma unroll
  for (int i = 0; i < 2; ++i) {
    const uint16_t* base = &lAp[(wmi * 32 + i * 16 + l15) * LDSW];
    a[i].h[0] = *(const U16x8*)&base[hi * 8];
    a[i].h[1] = *(const U16x8*)&base[16 + hi * 8];
  }
#pragma unroll
  for (int j = 0; j < 4; ++j) {
    const uint16_t* base = &lBp[(wni * 64 + j * 16 + l15) * LDSW];
    b[j].h[0] = *(const U16x8*)&base[hi * 8];
    b[j].h[1] = *(const U16x8*)&base[16 + hi * 8];
  }
#pragma unroll
  for (int i = 0; i < 2; ++i)
#pragma unroll
    for (int j = 0; j < 4; ++j)
      acc[i][j] = __builtin_amdgcn_wmma_f32_16x16x32_bf16(
          false, a[i].v, false, b[j].v, (short)0, acc[i][j], false, false);
}

// C[M,N] = act(A[M,K] @ W^T + bias). A: bf16 [M, lda]; W: bf16 [N, K] (pre-transposed).
// Optional f32 (stride ld32) and/or bf16 (stride ld16) outputs.
// Requires: M % 128 == 0, N % 128 == 0, K % 32 == 0.
__global__ __launch_bounds__(256)
void wmma_gemm_bias_act(const uint16_t* __restrict__ Abf,
                        const uint16_t* __restrict__ Wt,
                        const float* __restrict__ bias,
                        float* __restrict__ c32,
                        uint16_t* __restrict__ c16,
                        int M, int N, int K, int lda,
                        int ld32, int ld16, int act)  // act: 0=relu, 1=silu
{
  __shared__ __align__(16) uint16_t lA[2][BM * LDSW];
  __shared__ __align__(16) uint16_t lB[2][BN * LDSW];

  const int tid  = threadIdx.x;
  const int lane = tid & 31;
  const int wave = tid >> 5;
  const int wmi  = wave & 3;          // 4 waves along M (each 32 rows)
  const int wni  = wave >> 2;         // 2 waves along N (each 64 cols)
  const int l15  = lane & 15;
  const int hi   = lane >> 4;

  const int blockM = blockIdx.x * BM;
  const int blockN = blockIdx.y * BN;

  const v8f vzero = {0.f, 0.f, 0.f, 0.f, 0.f, 0.f, 0.f, 0.f};
  v8f acc[2][4];
#pragma unroll
  for (int i = 0; i < 2; ++i)
#pragma unroll
    for (int j = 0; j < 4; ++j) acc[i][j] = vzero;

#if HAVE_TDM
  // -------- async TDM path: double-buffered LDS, DMA overlaps WMMA ---------
  const uint32_t ldsA0 = (uint32_t)(uintptr_t)&lA[0][0];
  const uint32_t ldsA1 = (uint32_t)(uintptr_t)&lA[1][0];
  const uint32_t ldsB0 = (uint32_t)(uintptr_t)&lB[0][0];
  const uint32_t ldsB1 = (uint32_t)(uintptr_t)&lB[1][0];
  const uint16_t* Ablk = Abf + (size_t)blockM * lda;
  const uint16_t* Wblk = Wt  + (size_t)blockN * K;
  const int nk = K / BK;

  if (wave == 0) {
    tdm_load_tile(ldsA0, Ablk, lda, M, lda);
    tdm_load_tile(ldsB0, Wblk, K,  N, K);
  }
  for (int t = 0; t < nk; ++t) {
    const int cur = t & 1;
    if (wave == 0) {
      if (t + 1 < nk) {
        const int kt = (t + 1) * BK;
        tdm_load_tile(cur ? ldsA0 : ldsA1, Ablk + kt, lda - kt, M, lda);
        tdm_load_tile(cur ? ldsB0 : ldsB1, Wblk + kt, K  - kt, N, K);
        wait_tensorcnt<2>();           // tile t complete (tile t+1 in flight)
      } else {
        wait_tensorcnt<0>();
      }
    }
    __syncthreads();                   // tile t visible to all waves
    compute_tile(lA[cur], lB[cur], acc, wmi, wni, l15, hi);
    __syncthreads();                   // done reading before buffer is re-filled
  }
#else
  // -------- fallback: synchronous cooperative loads (round-1 scheme) -------
  const int row0 = tid >> 2;           // 0..63
  const int row1 = row0 + 64;          // 64..127
  const int c8   = (tid & 3) * 8;
  const size_t aoff0 = (size_t)(blockM + row0) * lda;
  const size_t aoff1 = (size_t)(blockM + row1) * lda;
  const size_t boff0 = (size_t)(blockN + row0) * K;
  const size_t boff1 = (size_t)(blockN + row1) * K;
  for (int kt = 0; kt < K; kt += BK) {
    __syncthreads();
    *(U16x8*)&lA[0][row0 * LDSW + c8] = *(const U16x8*)&Abf[aoff0 + kt + c8];
    *(U16x8*)&lA[0][row1 * LDSW + c8] = *(const U16x8*)&Abf[aoff1 + kt + c8];
    *(U16x8*)&lB[0][row0 * LDSW + c8] = *(const U16x8*)&Wt[boff0 + kt + c8];
    *(U16x8*)&lB[0][row1 * LDSW + c8] = *(const U16x8*)&Wt[boff1 + kt + c8];
    if (kt + BK < K) {
      __builtin_prefetch(&Abf[aoff0 + kt + BK + c8], 0, 1);
      __builtin_prefetch(&Wt[boff0 + kt + BK + c8], 0, 1);
    }
    __syncthreads();
    compute_tile(lA[0], lB[0], acc, wmi, wni, l15, hi);
  }
#endif

  // Epilogue: C/D layout — VGPR r, lane L -> row = r + (L>=16 ? 8 : 0), col = L&15.
#pragma unroll
  for (int i = 0; i < 2; ++i) {
    const int rbase = blockM + wmi * 32 + i * 16 + hi * 8;
#pragma unroll
    for (int j = 0; j < 4; ++j) {
      const int col = blockN + wni * 64 + j * 16 + l15;
      const float bv = bias ? bias[col] : 0.0f;
#pragma unroll
      for (int r = 0; r < 8; ++r) {
        float v = acc[i][j][r] + bv;
        if (act == 0) v = fmaxf(v, 0.0f);
        else          v = v / (1.0f + __expf(-v));   // silu
        const int row = rbase + r;
        if (c32) c32[(size_t)row * ld32 + col] = v;
        if (c16) c16[(size_t)row * ld16 + col] = f2bf(v);
      }
    }
  }
}

// ---------------------------------------------------------------------------
// LayerNorm over rows of length 1024: one wave32 per row, channel = row % 3.
__global__ __launch_bounds__(256)
void ln_rows_1024(float* __restrict__ X, uint16_t* __restrict__ outbf,
                  const float* __restrict__ g0, const float* __restrict__ b0,
                  const float* __restrict__ g1, const float* __restrict__ b1,
                  const float* __restrict__ g2, const float* __restrict__ b2,
                  int rows)
{
  const int row = blockIdx.x * 8 + (threadIdx.x >> 5);
  const int lane = threadIdx.x & 31;
  if (row >= rows) return;
  float* xr = X + (size_t)row * 1024;
  float v[32];
  float s = 0.f;
#pragma unroll
  for (int i = 0; i < 32; ++i) { v[i] = xr[i * 32 + lane]; s += v[i]; }
#pragma unroll
  for (int o = 16; o; o >>= 1) s += __shfl_xor(s, o, 32);
  const float mean = s * (1.0f / 1024.0f);
  float vs = 0.f;
#pragma unroll
  for (int i = 0; i < 32; ++i) { const float d = v[i] - mean; vs += d * d; }
#pragma unroll
  for (int o = 16; o; o >>= 1) vs += __shfl_xor(vs, o, 32);
  const float rstd = rsqrtf(vs * (1.0f / 1024.0f) + 1e-3f);
  const int ch = row % 3;
  const float* G  = (ch == 0) ? g0 : (ch == 1) ? g1 : g2;
  const float* Bt = (ch == 0) ? b0 : (ch == 1) ? b1 : b2;
  uint16_t* ob = outbf + (size_t)row * 1024;
#pragma unroll
  for (int i = 0; i < 32; ++i) {
    const int c = i * 32 + lane;
    const float y = G[c] * (v[i] - mean) * rstd + Bt[c];
    xr[c] = y;
    ob[c] = f2bf(y);
  }
}

// y[b,i,:] = sum_j Adj[b,i,j] * x[b,j,:]   (bf16 in/out, f32 math)
__global__ void mix3_kernel(const float* __restrict__ Adj,
                            const uint16_t* __restrict__ x,
                            uint16_t* __restrict__ y, int B, int d)
{
  const int gid = blockIdx.x * blockDim.x + threadIdx.x;
  if (gid >= B * d) return;
  const int b = gid / d, k = gid - b * d;
  const float* A = Adj + (size_t)b * 9;
  const size_t base = (size_t)b * 3 * d + k;
  const float x0 = bf2f(x[base]);
  const float x1 = bf2f(x[base + d]);
  const float x2 = bf2f(x[base + 2 * d]);
  y[base]         = f2bf(A[0] * x0 + A[1] * x1 + A[2] * x2);
  y[base + d]     = f2bf(A[3] * x0 + A[4] * x1 + A[5] * x2);
  y[base + 2 * d] = f2bf(A[6] * x0 + A[7] * x1 + A[8] * x2);
}

// out[b,:] = emb_bf16[t[b],:]  (embedding gather, makes every GEMM TDM-friendly)
__global__ void gather_rows_kernel(const uint16_t* __restrict__ emb,
                                   const int* __restrict__ t,
                                   uint16_t* __restrict__ out, int B, int d)
{
  const int gid = blockIdx.x * blockDim.x + threadIdx.x;
  if (gid >= B * d) return;
  const int b = gid / d, k = gid - b * d;
  out[gid] = emb[(size_t)t[b] * d + k];
}

// f32 [M,Kin] -> bf16 [M,Kpad] (zero padded K)
__global__ void cvt_pad_rows(const float* __restrict__ in, uint16_t* __restrict__ out,
                             int M, int Kin, int Kpad)
{
  const long long gid = (long long)blockIdx.x * blockDim.x + threadIdx.x;
  if (gid >= (long long)M * Kpad) return;
  const int r = (int)(gid / Kpad), k = (int)(gid % Kpad);
  out[gid] = (k < Kin) ? f2bf(in[(size_t)r * Kin + k]) : (uint16_t)0;
}

// f32 W[K,N] -> bf16 W^T[N,Kpad] (zero padded K)
__global__ void cvt_weight_T(const float* __restrict__ in, uint16_t* __restrict__ out,
                             int K, int N, int Kpad)
{
  const long long gid = (long long)blockIdx.x * blockDim.x + threadIdx.x;
  if (gid >= (long long)N * Kpad) return;
  const int n = (int)(gid / Kpad), k = (int)(gid % Kpad);
  out[gid] = (k < K) ? f2bf(in[(size_t)k * N + n]) : (uint16_t)0;
}

__global__ void sqdiff_kernel(const float* __restrict__ a, const float* __restrict__ b,
                              float* __restrict__ o, long long n)
{
  const long long i = (long long)blockIdx.x * blockDim.x + threadIdx.x;
  if (i >= n) return;
  const float d = a[i] - b[i];
  o[i] = d * d;
}

// br 0: o = 0.5*(X-Xh)^2 ; br 1: o += 0.5*(X-Xh)^2
__global__ void xloss_kernel(const float* __restrict__ X, const float* __restrict__ Xh,
                             float* __restrict__ o, long long n, int accumulate)
{
  const long long i = (long long)blockIdx.x * blockDim.x + threadIdx.x;
  if (i >= n) return;
  const float d = X[i] - Xh[i];
  const float v = 0.5f * d * d;
  o[i] = accumulate ? (o[i] + v) : v;
}

// ---------------------------------------------------------------------------
extern "C" void kernel_launch(void* const* d_in, const int* in_sizes, int n_in,
                              void* d_out, int out_size, void* d_ws, size_t ws_size,
                              hipStream_t stream)
{
  (void)in_sizes; (void)n_in; (void)out_size; (void)ws_size;

  const int B = 16384, D = 1024, DG = 772, DC = 100, NTT = 128;
  const int DGp = 800, DCp = 128;       // K padded to multiples of 32
  const int R = 3 * B;                  // 49152 rows for GCN layers

  const int*   t1 = (const int*)d_in[0];
  const float* g1 = (const float*)d_in[1];
  const float* c1 = (const float*)d_in[2];
  const float* A1 = (const float*)d_in[3];
  const int*   t2 = (const int*)d_in[4];
  const float* g2 = (const float*)d_in[5];
  const float* c2 = (const float*)d_in[6];
  const float* A2 = (const float*)d_in[7];
  const float* embed = (const float*)d_in[8];
  const float* Wt = (const float*)d_in[9];  const float* bt = (const float*)d_in[10];
  const float* Wg = (const float*)d_in[11]; const float* bg = (const float*)d_in[12];
  const float* Wc = (const float*)d_in[13]; const float* bc = (const float*)d_in[14];
  const float* lntg = (const float*)d_in[15]; const float* lntb = (const float*)d_in[16];
  const float* lngg = (const float*)d_in[17]; const float* lngb = (const float*)d_in[18];
  const float* lncg = (const float*)d_in[19]; const float* lncb = (const float*)d_in[20];
  const float* eW0 = (const float*)d_in[21]; const float* eb0 = (const float*)d_in[22];
  const float* eW1 = (const float*)d_in[23]; const float* eb1 = (const float*)d_in[24];
  const float* eW2 = (const float*)d_in[25]; const float* eb2 = (const float*)d_in[26];
  const float* dW0 = (const float*)d_in[27]; const float* db0 = (const float*)d_in[28];
  const float* dW1 = (const float*)d_in[29]; const float* db1 = (const float*)d_in[30];
  const float* dW2 = (const float*)d_in[31]; const float* db2 = (const float*)d_in[32];

  char* wsb = (char*)d_ws;
  size_t off = 0;
  auto alloc = [&](size_t bytes) -> void* {
    void* p = wsb + off;
    off = (off + bytes + 255) & ~(size_t)255;
    return p;
  };

  // bf16 weights, transposed [N][Kpad]
  uint16_t* wtb = (uint16_t*)alloc((size_t)1024 * 1024 * 2);
  uint16_t* wgb = (uint16_t*)alloc((size_t)1024 * DGp  * 2);
  uint16_t* wcb = (uint16_t*)alloc((size_t)1024 * DCp  * 2);
  uint16_t* e0b = (uint16_t*)alloc((size_t)1024 * 1024 * 2);
  uint16_t* e1b = (uint16_t*)alloc((size_t)512  * 1024 * 2);
  uint16_t* e2b = (uint16_t*)alloc((size_t)256  * 512  * 2);
  uint16_t* d0b = (uint16_t*)alloc((size_t)512  * 256  * 2);
  uint16_t* d1b = (uint16_t*)alloc((size_t)1024 * 512  * 2);
  uint16_t* d2b = (uint16_t*)alloc((size_t)1024 * 1024 * 2);
  uint16_t* emb = (uint16_t*)alloc((size_t)NTT * D * 2);
  uint16_t* tebf = (uint16_t*)alloc((size_t)B * D * 2);
  uint16_t* gbf  = (uint16_t*)alloc((size_t)B * DGp * 2);
  uint16_t* cbf  = (uint16_t*)alloc((size_t)B * DCp * 2);
  float* Xf0 = (float*)alloc((size_t)R * 1024 * 4);
  float* Xf1 = (float*)alloc((size_t)R * 1024 * 4);
  float* Zf0 = (float*)alloc((size_t)R * 256 * 4);
  float* Zf1 = (float*)alloc((size_t)R * 256 * 4);
  float* Xh  = (float*)alloc((size_t)R * 1024 * 4);
  uint16_t* actA = (uint16_t*)alloc((size_t)R * 1024 * 2);
  uint16_t* actB = (uint16_t*)alloc((size_t)R * 1024 * 2);

  auto cvtW = [&](const float* src, uint16_t* dst, int K, int N, int Kpad) {
    const long long n = (long long)N * Kpad;
    cvt_weight_T<<<(unsigned)((n + 255) / 256), 256, 0, stream>>>(src, dst, K, N, Kpad);
  };
  cvtW(Wt,  wtb, 1024, 1024, 1024);
  cvtW(Wg,  wgb, 772,  1024, DGp);
  cvtW(Wc,  wcb, 100,  1024, DCp);
  cvtW(eW0, e0b, 1024, 1024, 1024);
  cvtW(eW1, e1b, 1024, 512,  1024);
  cvtW(eW2, e2b, 512,  256,  512);
  cvtW(dW0, d0b, 256,  512,  256);
  cvtW(dW1, d1b, 512,  1024, 512);
  cvtW(dW2, d2b, 1024, 1024, 1024);
  cvt_pad_rows<<<(NTT * D + 255) / 256, 256, 0, stream>>>(embed, emb, NTT, D, D);

  auto gemm = [&](const uint16_t* A, const uint16_t* W, const float* bias,
                  float* o32, uint16_t* o16,
                  int M, int N, int K, int lda, int ld32, int ld16, int act) {
    dim3 grid((unsigned)(M / BM), (unsigned)(N / BN));
    wmma_gemm_bias_act<<<grid, 256, 0, stream>>>(A, W, bias, o32, o16,
                                                 M, N, K, lda, ld32, ld16, act);
  };
  auto mix = [&](const float* Aj, const uint16_t* x, uint16_t* y, int d) {
    const int n = B * d;
    mix3_kernel<<<(n + 255) / 256, 256, 0, stream>>>(Aj, x, y, B, d);
  };

  for (int br = 0; br < 2; ++br) {
    const int*   t  = br ? t2 : t1;
    const float* g  = br ? g2 : g1;
    const float* c  = br ? c2 : c1;
    const float* Aj = br ? A2 : A1;
    float* X = br ? Xf1 : Xf0;
    float* Z = br ? Zf1 : Zf0;

    gather_rows_kernel<<<(B * D + 255) / 256, 256, 0, stream>>>(emb, t, tebf, B, D);
    { const long long n = (long long)B * DGp;
      cvt_pad_rows<<<(unsigned)((n + 255) / 256), 256, 0, stream>>>(g, gbf, B, DG, DGp); }
    { const long long n = (long long)B * DCp;
      cvt_pad_rows<<<(unsigned)((n + 255) / 256), 256, 0, stream>>>(c, cbf, B, DC, DCp); }

    // Branch projections -> X f32 [B,3,1024] (row stride 3072, channel offsets)
    gemm(tebf, wtb, bt, X + 0,    nullptr, B, 1024, 1024, 1024, 3072, 0, 0);
    gemm(gbf,  wgb, bg, X + 1024, nullptr, B, 1024, DGp,  DGp,  3072, 0, 0);
    gemm(cbf,  wcb, bc, X + 2048, nullptr, B, 1024, DCp,  DCp,  3072, 0, 0);

    // LayerNorm (in place f32) + bf16 copy for encoder input
    ln_rows_1024<<<R / 8, 256, 0, stream>>>(X, actA, lntg, lntb, lngg, lngb, lncg, lncb, R);

    // Encoder (relu)
    mix(Aj, actA, actB, 1024);
    gemm(actB, e0b, eb0, nullptr, actA, R, 1024, 1024, 1024, 0, 1024, 0);
    mix(Aj, actA, actB, 1024);
    gemm(actB, e1b, eb1, nullptr, actA, R, 512,  1024, 1024, 0, 512,  0);
    mix(Aj, actA, actB, 512);
    gemm(actB, e2b, eb2, Z,       actA, R, 256,  512,  512,  256, 256, 0);

    // Decoder (silu)
    mix(Aj, actA, actB, 256);
    gemm(actB, d0b, db0, nullptr, actA, R, 512,  256,  256,  0, 512,  1);
    mix(Aj, actA, actB, 512);
    gemm(actB, d1b, db1, nullptr, actA, R, 1024, 512,  512,  0, 1024, 1);
    mix(Aj, actA, actB, 1024);
    gemm(actB, d2b, db2, Xh, nullptr,   R, 1024, 1024, 1024, 1024, 0, 1);

    // X-loss partial: 0.5*(X - Xh)^2, accumulated over branches
    const long long nx = (long long)R * 1024;
    xloss_kernel<<<(unsigned)((nx + 255) / 256), 256, 0, stream>>>(
        X, Xh, (float*)d_out + (long long)R * 256, nx, br);
  }

  // Z-loss
  const long long nz = (long long)R * 256;
  sqdiff_kernel<<<(unsigned)((nz + 255) / 256), 256, 0, stream>>>(
      Zf0, Zf1, (float*)d_out, nz);
}
